// CGIM_22806276341864
// MI455X (gfx1250) — matrix operand order repaired and checked by reading, not compiled, not run on hardware
//
#include <hip/hip_runtime.h>

#define HW   4096
#define BB   8
#define ALD  40     // A-tile / generic LDS row stride (ushorts): 80B = 5x16B, conflict-free for frag reads
#define BSTP 136    // proj B-tile [k][n] row stride (128 + 8)
#define XWST 264    // xw   B-tile [e][n] row stride (256 + 8)

typedef __attribute__((ext_vector_type(16))) __bf16    v16bf;
typedef __attribute__((ext_vector_type(8)))  float     v8f;
typedef __attribute__((ext_vector_type(4)))  int       v4i;
typedef __attribute__((ext_vector_type(8)))  int       v8i;
typedef __attribute__((ext_vector_type(4)))  unsigned  v4u;

union Frag16 { unsigned u[8]; v16bf v; };
union FragAcc { v8f v; float f[8]; };

__device__ __forceinline__ unsigned short f2bf(float x) {
  union { float f; unsigned u; } c; c.f = x;
  unsigned u = c.u;
  u += 0x7FFFu + ((u >> 16) & 1u);     // round-to-nearest-even
  return (unsigned short)(u >> 16);
}
__device__ __forceinline__ float bf2f(unsigned short h) {
  union { unsigned u; float f; } c; c.u = ((unsigned)h) << 16;
  return c.f;
}

// ---------------- async global->LDS copy (CDNA5 GLOBAL_LOAD_ASYNC_TO_LDS_B128) ----------------
#if __has_builtin(__builtin_amdgcn_global_load_async_to_lds_b128) && __has_builtin(__builtin_amdgcn_s_wait_asynccnt)
#define USE_ASYNC_COPY 1
#else
#define USE_ASYNC_COPY 0
#endif

// ---------------- Tensor Data Mover (CDNA5 TENSOR_LOAD_TO_LDS, TENSORcnt) ----------------
#if USE_ASYNC_COPY && __has_builtin(__builtin_amdgcn_tensor_load_to_lds) && __has_builtin(__builtin_amdgcn_s_wait_tensorcnt)
#define USE_TDM 1
#else
#define USE_TDM 0
#endif

// both pointers must be 16B aligned; copies 16 bytes per lane
__device__ __forceinline__ void async_copy_b128(unsigned short* lds_dst, const unsigned short* gsrc) {
#if USE_ASYNC_COPY
  __builtin_amdgcn_global_load_async_to_lds_b128(
      (__attribute__((address_space(1))) v4i*)(gsrc),
      (__attribute__((address_space(3))) v4i*)(lds_dst), 0, 0);
#else
  *(uint4*)lds_dst = *(const uint4*)gsrc;
#endif
}
__device__ __forceinline__ void async_wait0() {
#if USE_ASYNC_COPY
  __builtin_amdgcn_s_wait_asynccnt(0);
#endif
}
__device__ __forceinline__ void async_wait4() {
#if USE_ASYNC_COPY
  __builtin_amdgcn_s_wait_asynccnt(4);
#endif
}
__device__ __forceinline__ void tensor_wait0() {
#if USE_TDM
  __builtin_amdgcn_s_wait_tensorcnt(0);
#endif
}

#if USE_TDM
__device__ __forceinline__ unsigned lds_addr_of(const void* p) {
  return (unsigned)(unsigned long long)(__attribute__((address_space(3))) const void*)p;
}
// 2D TDM tile load: 32 rows of 128 dwords (256 bf16), global row stride HW elems,
// LDS row stride XWST ushorts (pad 4 dwords after every 128 dwords).
__device__ __forceinline__ void tdm_load_tile_256(unsigned short* lds_dst, const unsigned short* gsrc) {
  unsigned long long ga = (unsigned long long)(const void*)gsrc;
  v4u g0;
  g0.x = 1u;                                        // count=1, user descriptor, no gather
  g0.y = lds_addr_of(lds_dst);                      // lds_addr (bytes)
  g0.z = (unsigned)(ga & 0xFFFFFFFFu);              // global_addr[31:0]
  g0.w = (unsigned)((ga >> 32) & 0x01FFFFFFu)       // global_addr[56:32]
       | (2u << 30);                                // type = 2 ("image")
  // group1: data_size=4B(code2), pad_enable, pad_interval=128dw(code6), pad_amount=4dw(code3)
  int w0 = (int)((2u << 16) | (1u << 20) | (6u << 22) | (3u << 25));
  int w1 = (int)(128u << 16);   // tensor_dim0 = 128 dwords (bits 79:48, low half)
  int w2 = (int)(32u << 16);    // tensor_dim0 hi=0 | tensor_dim1 = 32 (bits 111:80, low half)
  int w3 = (int)(128u << 16);   // tensor_dim1 hi=0 | tile_dim0 = 128 dwords
  int w4 = (int)32;             // tile_dim1 = 32, tile_dim2 = 0
  int w5 = (int)2048;           // tensor_dim0_stride = HW*2B/4B = 2048 dwords (bits 207:160, low 32)
  v8i g1 = { w0, w1, w2, w3, w4, w5, 0, 0 };
  v4i g2 = { 0, 0, 0, 0 };
  v4i g3 = { 0, 0, 0, 0 };
  v8i g4 = { 0, 0, 0, 0, 0, 0, 0, 0 };
  __builtin_amdgcn_tensor_load_to_lds(g0, g1, g2, g3, g4, 0);
}
#endif

// ---- WMMA fragment loads (wave32; this code's k order k = 16*(lane/16) + s, same in A and B) ----
// A-matrix 16x32 bf16 from LDS [m][k], row stride ALD
__device__ __forceinline__ void load_frag_a(const unsigned short* sm, int row0, Frag16& f) {
  int lane = threadIdx.x & 31;
  int half = lane >> 4;
  const unsigned short* rp = sm + (row0 + (lane & 15)) * ALD;
#pragma unroll
  for (int v = 0; v < 8; ++v) {
    int k = half * 16 + 2 * v;
    f.u[v] = *(const unsigned*)(rp + k);
  }
}
// B-matrix 32x16 bf16 from LDS stored transposed [n][k], row stride ALD (conv path)
__device__ __forceinline__ void load_frag_b_tr(const unsigned short* sm, int col0, Frag16& f) {
  int lane = threadIdx.x & 31;
  const unsigned short* cp = sm + (col0 + (lane & 15)) * ALD + ((lane >> 4) << 4);
#pragma unroll
  for (int v = 0; v < 8; ++v) f.u[v] = *(const unsigned*)(cp + 2 * v);
}
// B-matrix 32x16 bf16 from LDS stored natural [k][n], row stride ST (async-copied tiles)
template <int ST>
__device__ __forceinline__ void load_frag_b_kn(const unsigned short* sm, int col0, Frag16& f) {
  int lane = threadIdx.x & 31;
  const unsigned short* p = sm + (((lane >> 4) << 4)) * ST + col0 + (lane & 15);
#pragma unroll
  for (int v = 0; v < 8; ++v) {
    unsigned lo = p[(2 * v) * ST];
    unsigned hi = p[(2 * v + 1) * ST];
    f.u[v] = lo | (hi << 16);
  }
}

__device__ __forceinline__ void wmma_bf16(FragAcc& acc, const Frag16& a, const Frag16& b) {
  acc.v = __builtin_amdgcn_wmma_f32_16x16x32_bf16(false, a.v, false, b.v,
                                                  (short)0, acc.v, false, false);
}

// ---------------------------------------------------------------- converts
__global__ __launch_bounds__(256) void convert_kernel(const float* __restrict__ src,
                                                      unsigned short* __restrict__ dst, long n) {
  long i = (long)blockIdx.x * blockDim.x + threadIdx.x;
  long stride = (long)gridDim.x * blockDim.x;
  for (; i < n; i += stride) dst[i] = f2bf(src[i]);
}

__global__ __launch_bounds__(256) void convert_batched_kernel(const float* __restrict__ src,
                                                              unsigned short* __restrict__ dst,
                                                              long chunk, long dstStride, int nb) {
  long total = chunk * nb;
  long i = (long)blockIdx.x * blockDim.x + threadIdx.x;
  long stride = (long)gridDim.x * blockDim.x;
  for (; i < total; i += stride) {
    long b = i / chunk, r = i - b * chunk;
    dst[b * dstStride + r] = f2bf(src[i]);
  }
}

// ------------------------------------------------- 1x1 projection GEMM (WMMA, async double-buffered)
// Out[b][m][n] = sum_k W[m][k] * X[b][k][n] + bias[m]   (M=256, N=4096, K=Kdim)
#define TM 128
#define TN 128
#define TK 32
__global__ __launch_bounds__(256) void proj_gemm_kernel(
    const unsigned short* __restrict__ W,     // [256][Kdim] bf16
    const unsigned short* __restrict__ X,     // [B][512][HW] bf16 (XCAT)
    const float* __restrict__ bias,
    unsigned short* __restrict__ Out,         // [B][256][HW] bf16
    int Kdim, long srcOff)
{
  __shared__ alignas(16) unsigned short Asm[2][TM * ALD];
  __shared__ alignas(16) unsigned short Bsm[2][TK * BSTP];

  const int b  = blockIdx.z;
  const int n0 = blockIdx.x * TN;
  const int m0 = blockIdx.y * TM;
  const int tid = threadIdx.x;
  const int wave = tid >> 5;
  const int wm = (wave & 1) * 64;
  const int wn = (wave >> 1) * 32;

  const unsigned short* Xb = X + (long)b * 512 * HW + srcOff;

  FragAcc acc[4][2];
#pragma unroll
  for (int i = 0; i < 4; ++i)
#pragma unroll
    for (int j = 0; j < 2; ++j)
#pragma unroll
      for (int e = 0; e < 8; ++e) acc[i][j].f[e] = 0.f;

  // per-thread tile-copy: 2 async b128 for A, 2 for B => 4 async instr per wave per tile
  auto loadTile = [&](int k0, int buf) {
    int m = tid & 127;
    int halfk = (tid >> 7) << 4;
    const unsigned short* asrc = W + (long)(m0 + m) * Kdim + k0 + halfk;
    unsigned short* ad = Asm[buf] + m * ALD + halfk;
    async_copy_b128(ad, asrc);
    async_copy_b128(ad + 8, asrc + 8);
#pragma unroll
    for (int rep = 0; rep < 2; ++rep) {
      int g = tid + rep * 256;
      int r = g >> 4;
      int col8 = g & 15;
      async_copy_b128(Bsm[buf] + r * BSTP + col8 * 8,
                      Xb + (long)(k0 + r) * HW + n0 + col8 * 8);
    }
  };

  const int nk = Kdim / TK;
  loadTile(0, 0);
  for (int t = 0; t < nk; ++t) {
    const int buf = t & 1;
    if (t + 1 < nk) { loadTile((t + 1) * TK, buf ^ 1); async_wait4(); }
    else            { async_wait0(); }
    __syncthreads();

    Frag16 af[4], bfr[2];
#pragma unroll
    for (int i = 0; i < 4; ++i) load_frag_a(Asm[buf], wm + i * 16, af[i]);
#pragma unroll
    for (int j = 0; j < 2; ++j) load_frag_b_kn<BSTP>(Bsm[buf], wn + j * 16, bfr[j]);
#pragma unroll
    for (int i = 0; i < 4; ++i)
#pragma unroll
      for (int j = 0; j < 2; ++j) wmma_bf16(acc[i][j], af[i], bfr[j]);
    __syncthreads();
  }

  const int lane = tid & 31, ln = lane & 15, lh = lane >> 4;
  unsigned short* Ob = Out + (long)b * 256 * HW;
#pragma unroll
  for (int i = 0; i < 4; ++i)
#pragma unroll
    for (int j = 0; j < 2; ++j) {
      int n = n0 + wn + j * 16 + ln;
#pragma unroll
      for (int e = 0; e < 8; ++e) {
        int m = m0 + wm + i * 16 + e + lh * 8;
        Ob[(long)m * HW + n] = f2bf(acc[i][j].f[e] + bias[m]);
      }
    }
}

// ------------------------------------------------- softmax over d (32 channels)
__global__ __launch_bounds__(256) void softmax_d_kernel(
    const unsigned short* __restrict__ Qb,   // [B][256][HW]
    unsigned short* __restrict__ Qs)         // [B][8][HW][32]
{
  long g = (long)blockIdx.x * blockDim.x + threadIdx.x;   // (b*8+h)*HW + n
  int n = (int)(g & (HW - 1));
  int h = (int)((g >> 12) & 7);
  int b = (int)(g >> 15);
  const unsigned short* src = Qb + ((long)b * 256 + h * 32) * HW + n;
  float vals[32];
  float mx = -1e30f;
#pragma unroll
  for (int d = 0; d < 32; ++d) { vals[d] = bf2f(src[(long)d * HW]); mx = fmaxf(mx, vals[d]); }
  float s = 0.f;
#pragma unroll
  for (int d = 0; d < 32; ++d) { vals[d] = __expf(vals[d] - mx); s += vals[d]; }
  float inv = 1.f / s;
  unsigned short* dst = Qs + g * 32;
#pragma unroll
  for (int d = 0; d < 32; ++d) dst[d] = f2bf(vals[d] * inv);
}

// ------------------------------------------------- softmax over HW, in place
__global__ __launch_bounds__(256) void softmax_hw_kernel(unsigned short* __restrict__ K1,
                                                         unsigned short* __restrict__ K2)
{
  unsigned short* row = ((blockIdx.y == 0) ? K1 : K2) + (long)blockIdx.x * HW;
  __shared__ float red[256];
  int t = threadIdx.x;
  float v[16]; float mx = -1e30f;
#pragma unroll
  for (int i = 0; i < 16; ++i) { v[i] = bf2f(row[t + i * 256]); mx = fmaxf(mx, v[i]); }
  red[t] = mx; __syncthreads();
  for (int s = 128; s > 0; s >>= 1) { if (t < s) red[t] = fmaxf(red[t], red[t + s]); __syncthreads(); }
  mx = red[0]; __syncthreads();
  float sum = 0.f;
#pragma unroll
  for (int i = 0; i < 16; ++i) { v[i] = __expf(v[i] - mx); sum += v[i]; }
  red[t] = sum; __syncthreads();
  for (int s = 128; s > 0; s >>= 1) { if (t < s) red[t] += red[t + s]; __syncthreads(); }
  float inv = 1.f / red[0];
#pragma unroll
  for (int i = 0; i < 16; ++i) row[t + i * 256] = f2bf(v[i] * inv);
}

// ------------------------------------------------- Att = Ks (32xHW) @ Qs (HWx32)
__global__ __launch_bounds__(128) void att_kernel(
    const unsigned short* __restrict__ Ks1, const unsigned short* __restrict__ Ks2,
    const unsigned short* __restrict__ Qs,
    unsigned short* __restrict__ Att)        // [2][64][32*32] bf16
{
  __shared__ alignas(16) unsigned short Asm[4][32 * ALD];
  __shared__ alignas(16) unsigned short Bsm[4][32 * ALD];   // [k][e] natural layout
  __shared__ float red[4][1024];

  int bh = blockIdx.x, br = blockIdx.y;
  int b = bh >> 3, h = bh & 7;
  int tid = threadIdx.x, wave = tid >> 5, lane = tid & 31;
  const unsigned short* Kbase = ((br == 0) ? Ks1 : Ks2) + ((long)b * 256 + h * 32) * HW;
  const unsigned short* Qbase = Qs + (long)bh * HW * 32;

  FragAcc acc[2][2];
#pragma unroll
  for (int i = 0; i < 2; ++i)
#pragma unroll
    for (int j = 0; j < 2; ++j)
#pragma unroll
      for (int e = 0; e < 8; ++e) acc[i][j].f[e] = 0.f;

  for (int kb = wave * 1024; kb < (wave + 1) * 1024; kb += 32) {
    // A: row d=lane, 32 contiguous k from Ks -> Asm[wave][lane][*]
#pragma unroll
    for (int c = 0; c < 4; ++c)
      async_copy_b128(Asm[wave] + lane * ALD + c * 8, Kbase + (long)lane * HW + kb + c * 8);
    // B: row n=kb+lane of Qs (32 contiguous e) -> Bsm[wave][lane][*]  ([k][e] natural)
#pragma unroll
    for (int c = 0; c < 4; ++c)
      async_copy_b128(Bsm[wave] + lane * ALD + c * 8, Qbase + (long)(kb + lane) * 32 + c * 8);
    async_wait0();
    __syncthreads();

    Frag16 af[2], bfr[2];
    load_frag_a(Asm[wave], 0, af[0]);            load_frag_a(Asm[wave], 16, af[1]);
    load_frag_b_kn<ALD>(Bsm[wave], 0, bfr[0]);   load_frag_b_kn<ALD>(Bsm[wave], 16, bfr[1]);
#pragma unroll
    for (int i = 0; i < 2; ++i)
#pragma unroll
      for (int j = 0; j < 2; ++j) wmma_bf16(acc[i][j], af[i], bfr[j]);
    __syncthreads();
  }

  int ln = lane & 15, lh = lane >> 4;
#pragma unroll
  for (int i = 0; i < 2; ++i)
#pragma unroll
    for (int j = 0; j < 2; ++j)
#pragma unroll
      for (int e = 0; e < 8; ++e)
        red[wave][(i * 16 + e + lh * 8) * 32 + (j * 16 + ln)] = acc[i][j].f[e];
  __syncthreads();
  for (int o = tid; o < 1024; o += 128) {
    float s = red[0][o] + red[1][o] + red[2][o] + red[3][o];
    Att[((long)br * 64 + bh) * 1024 + o] = f2bf(s);
  }
}

// ------------------------------------------------- Xw = Att @ V fused with mu*X+F -> bf16 fused buf
__global__ __launch_bounds__(256) void xw_fuse_kernel(
    const unsigned short* __restrict__ Att,
    const unsigned short* __restrict__ V1, const unsigned short* __restrict__ V2,
    const float* __restrict__ F1, const float* __restrict__ F2,
    const float* __restrict__ mu,
    unsigned short* __restrict__ Fused)      // [B][512][HW]
{
  __shared__ alignas(16) unsigned short AttSm[32 * ALD];     // [m][e]
  __shared__ alignas(16) unsigned short Bsm[32 * XWST];      // [e][n] natural

  int n0 = blockIdx.x * 256;
  int bh = blockIdx.y, br = blockIdx.z;
  int b = bh >> 3, h = bh & 7;
  int tid = threadIdx.x;
  const unsigned short* Vb = ((br == 0) ? V1 : V2) + ((long)b * 256 + h * 32) * HW;
  const float* Fb = ((br == 0) ? F1 : F2) + ((long)b * 256 + h * 32) * HW;
  const unsigned short* Ag = Att + ((long)br * 64 + bh) * 1024;

  if (tid < 128) {
    int row = tid >> 2, c = tid & 3;
    async_copy_b128(AttSm + row * ALD + c * 8, Ag + row * 32 + c * 8);
  }
#if USE_TDM
  // V tile: one TDM descriptor moves 32 rows x 256 bf16 (row stride HW) into Bsm with row pad
  if (tid == 0) tdm_load_tile_256(Bsm, Vb + n0);
  async_wait0();
  tensor_wait0();
#else
#pragma unroll
  for (int j = 0; j < 4; ++j) {
    int c = tid + 256 * j;
    int e = c >> 5, coloff = (c & 31) * 8;
    async_copy_b128(Bsm + e * XWST + coloff, Vb + (long)e * HW + n0 + coloff);
  }
  async_wait0();
#endif
  __syncthreads();

  int wave = tid >> 5, lane = tid & 31;
  int wn = wave * 32;
  Frag16 af[2], bfr[2];
  load_frag_a(AttSm, 0, af[0]); load_frag_a(AttSm, 16, af[1]);
  load_frag_b_kn<XWST>(Bsm, wn, bfr[0]); load_frag_b_kn<XWST>(Bsm, wn + 16, bfr[1]);
  FragAcc acc[2][2];
#pragma unroll
  for (int i = 0; i < 2; ++i)
#pragma unroll
    for (int j = 0; j < 2; ++j) {
#pragma unroll
      for (int e = 0; e < 8; ++e) acc[i][j].f[e] = 0.f;
      wmma_bf16(acc[i][j], af[i], bfr[j]);
    }

  float muv = mu[0];
  int ln = lane & 15, lh = lane >> 4;
  unsigned short* Ob = Fused + ((long)b * 512 + br * 256 + h * 32) * HW;
#pragma unroll
  for (int i = 0; i < 2; ++i)
#pragma unroll
    for (int j = 0; j < 2; ++j) {
      int n = n0 + wn + j * 16 + ln;
#pragma unroll
      for (int e = 0; e < 8; ++e) {
        int d = i * 16 + e + lh * 8;
        float val = muv * acc[i][j].f[e] + Fb[(long)d * HW + n];
        Ob[(long)d * HW + n] = f2bf(val);
      }
    }
}

// ------------------------------------------------- implicit-GEMM 3x3 conv + BN + ReLU
__global__ __launch_bounds__(256) void conv3x3_bn_relu_kernel(
    const unsigned short* __restrict__ Wc,     // [256][4608] bf16 (OIHW flat)
    const unsigned short* __restrict__ Fused,  // [B][512][HW] bf16
    const float* __restrict__ gamma, const float* __restrict__ beta,
    const float* __restrict__ rmean, const float* __restrict__ rvar,
    float* __restrict__ Out)                   // [B][256][HW] fp32
{
  __shared__ alignas(16) unsigned short Asm[TM * ALD];
  __shared__ alignas(16) unsigned short Bsm[TN * ALD];   // transposed [n][k]

  const int b  = blockIdx.z;
  const int n0 = blockIdx.x * TN;
  const int m0 = blockIdx.y * TM;
  const int tid = threadIdx.x;
  const int wave = tid >> 5;
  const int wm = (wave & 1) * 64;
  const int wn = (wave >> 1) * 32;
  const unsigned short* Fb = Fused + (long)b * 512 * HW;

  FragAcc acc[4][2];
#pragma unroll
  for (int i = 0; i < 4; ++i)
#pragma unroll
    for (int j = 0; j < 2; ++j)
#pragma unroll
      for (int e = 0; e < 8; ++e) acc[i][j].f[e] = 0.f;

  for (int tap = 0; tap < 9; ++tap) {
    const int dh = tap / 3 - 1, dw = tap % 3 - 1;
    for (int i0 = 0; i0 < 512; i0 += TK) {
      // A tile: Asm[m][kk] = Wc[m0+m][(i0+kk)*9 + tap]   (stride-9 gather)
      {
        int m = tid & 127;
        int halfk = (tid >> 7) << 4;
        const unsigned short* src = Wc + (long)(m0 + m) * 4608 + (long)(i0 + halfk) * 9 + tap;
        unsigned short* d = Asm + m * ALD + halfk;
#pragma unroll
        for (int kk = 0; kk < 16; ++kk) d[kk] = src[kk * 9];
      }
      // B tile: im2col of shifted fused rows, transposed [n][k]
#pragma unroll
      for (int rep = 0; rep < 2; ++rep) {
        int g = tid + rep * 256;
        int r = g >> 4;                 // cin within tile
        int col8 = g & 15;
        int nl = col8 * 8;
        int n = n0 + nl;
        int hq = n >> 6, wq = n & 63;
        int hh = hq + dh;
        bool hok = (hh >= 0) && (hh < 64);
        unsigned short* bd = Bsm + nl * ALD + r;
        long rowbase = (long)(i0 + r) * HW + (long)hh * 64;
        if (i0 + TK < 512) __builtin_prefetch(Fb + (long)(i0 + TK + r) * HW + (long)((hh >= 0) ? hh : 0) * 64 + wq, 0, 1);
#pragma unroll
        for (int i = 0; i < 8; ++i) {
          int ww = wq + i + dw;
          unsigned short vx = 0;
          if (hok && ww >= 0 && ww < 64) vx = Fb[rowbase + ww];
          bd[i * ALD] = vx;
        }
      }
      __syncthreads();

      Frag16 af[4], bfr[2];
#pragma unroll
      for (int i = 0; i < 4; ++i) load_frag_a(Asm, wm + i * 16, af[i]);
#pragma unroll
      for (int j = 0; j < 2; ++j) load_frag_b_tr(Bsm, wn + j * 16, bfr[j]);
#pragma unroll
      for (int i = 0; i < 4; ++i)
#pragma unroll
        for (int j = 0; j < 2; ++j) wmma_bf16(acc[i][j], af[i], bfr[j]);
      __syncthreads();
    }
  }

  const int lane = tid & 31, ln = lane & 15, lh = lane >> 4;
  float* Ob = Out + (long)b * 256 * HW;
#pragma unroll
  for (int i = 0; i < 4; ++i)
#pragma unroll
    for (int j = 0; j < 2; ++j) {
      int n = n0 + wn + j * 16 + ln;
#pragma unroll
      for (int e = 0; e < 8; ++e) {
        int m = m0 + wm + i * 16 + e + lh * 8;
        float inv = gamma[m] * rsqrtf(rvar[m] + 1e-5f);
        float y = acc[i][j].f[e] * inv + (beta[m] - rmean[m] * inv);
        Ob[(long)m * HW + n] = fmaxf(y, 0.f);
      }
    }
}

// ---------------------------------------------------------------- host launcher
extern "C" void kernel_launch(void* const* d_in, const int* in_sizes, int n_in,
                              void* d_out, int out_size, void* d_ws, size_t ws_size,
                              hipStream_t stream)
{
  (void)in_sizes; (void)n_in; (void)out_size; (void)ws_size;
  const float* F1  = (const float*)d_in[0];
  const float* F2  = (const float*)d_in[1];
  const float* Wq  = (const float*)d_in[2];
  const float* bq  = (const float*)d_in[3];
  const float* Wk1 = (const float*)d_in[4];
  const float* bk1 = (const float*)d_in[5];
  const float* Wv1 = (const float*)d_in[6];
  const float* bv1 = (const float*)d_in[7];
  const float* Wk2 = (const float*)d_in[8];
  const float* bk2 = (const float*)d_in[9];
  const float* Wv2 = (const float*)d_in[10];
  const float* bv2 = (const float*)d_in[11];
  const float* mu  = (const float*)d_in[12];
  const float* Wc  = (const float*)d_in[13];
  const float* gamma = (const float*)d_in[14];
  const float* beta  = (const float*)d_in[15];
  const float* rmean = (const float*)d_in[16];
  const float* rvar  = (const float*)d_in[17];
  float* out = (float*)d_out;

  char* p = (char*)d_ws;
  auto alloc = [&](size_t bytes) -> char* {
    char* r = p; p += (bytes + 255) & ~(size_t)255; return r;
  };
  unsigned short* XCAT = (unsigned short*)alloc((size_t)BB * 512 * HW * 2); // F1||F2 bf16, reused as fused conv input
  unsigned short* QB   = (unsigned short*)alloc((size_t)BB * 256 * HW * 2);
  unsigned short* QS   = (unsigned short*)alloc((size_t)BB * 8 * HW * 32 * 2);
  unsigned short* K1B  = (unsigned short*)alloc((size_t)BB * 256 * HW * 2);
  unsigned short* K2B  = (unsigned short*)alloc((size_t)BB * 256 * HW * 2);
  unsigned short* V1B  = (unsigned short*)alloc((size_t)BB * 256 * HW * 2);
  unsigned short* V2B  = (unsigned short*)alloc((size_t)BB * 256 * HW * 2);
  unsigned short* WQB  = (unsigned short*)alloc((size_t)256 * 512 * 2);
  unsigned short* WK1B = (unsigned short*)alloc((size_t)256 * 256 * 2);
  unsigned short* WV1B = (unsigned short*)alloc((size_t)256 * 256 * 2);
  unsigned short* WK2B = (unsigned short*)alloc((size_t)256 * 256 * 2);
  unsigned short* WV2B = (unsigned short*)alloc((size_t)256 * 256 * 2);
  unsigned short* WCB  = (unsigned short*)alloc((size_t)256 * 4608 * 2);
  unsigned short* ATT  = (unsigned short*)alloc((size_t)2 * 64 * 1024 * 2);

  // 1) fp32 -> bf16 converts
  convert_batched_kernel<<<2048, 256, 0, stream>>>(F1, XCAT, (long)256 * HW, (long)512 * HW, BB);
  convert_batched_kernel<<<2048, 256, 0, stream>>>(F2, XCAT + (long)256 * HW, (long)256 * HW, (long)512 * HW, BB);
  convert_kernel<<<512, 256, 0, stream>>>(Wq,  WQB,  (long)256 * 512);
  convert_kernel<<<256, 256, 0, stream>>>(Wk1, WK1B, (long)256 * 256);
  convert_kernel<<<256, 256, 0, stream>>>(Wv1, WV1B, (long)256 * 256);
  convert_kernel<<<256, 256, 0, stream>>>(Wk2, WK2B, (long)256 * 256);
  convert_kernel<<<256, 256, 0, stream>>>(Wv2, WV2B, (long)256 * 256);
  convert_kernel<<<2048, 256, 0, stream>>>(Wc, WCB, (long)256 * 4608);

  // 2) projection GEMMs (WMMA bf16, async double-buffered)
  dim3 gproj(HW / TN, 256 / TM, BB);
  proj_gemm_kernel<<<gproj, 256, 0, stream>>>(WQB,  XCAT, bq,  QB,  512, 0L);
  proj_gemm_kernel<<<gproj, 256, 0, stream>>>(WK1B, XCAT, bk1, K1B, 256, 0L);
  proj_gemm_kernel<<<gproj, 256, 0, stream>>>(WV1B, XCAT, bv1, V1B, 256, 0L);
  proj_gemm_kernel<<<gproj, 256, 0, stream>>>(WK2B, XCAT, bk2, K2B, 256, (long)256 * HW);
  proj_gemm_kernel<<<gproj, 256, 0, stream>>>(WV2B, XCAT, bv2, V2B, 256, (long)256 * HW);

  // 3) softmaxes
  softmax_d_kernel<<<(BB * 8 * HW) / 256, 256, 0, stream>>>(QB, QS);
  softmax_hw_kernel<<<dim3(BB * 256, 2), 256, 0, stream>>>(K1B, K2B);

  // 4) Att = Ks @ Qs (WMMA, K-split across 4 waves, async staged)
  att_kernel<<<dim3(64, 2), 128, 0, stream>>>(K1B, K2B, QS, ATT);

  // 5) Xw = Att @ V fused with mu*X + F, written in place over XCAT as conv input
  xw_fuse_kernel<<<dim3(HW / 256, 64, 2), 256, 0, stream>>>(ATT, V1B, V2B, F1, F2, mu, XCAT);

  // 6) 3x3 conv (implicit GEMM over 9 taps x 16 cin tiles) + BN + ReLU
  conv3x3_bn_relu_kernel<<<dim3(HW / TN, 256 / TM, BB), 256, 0, stream>>>(
      WCB, XCAT, gamma, beta, rmean, rvar, out);
}
